// DAT_68805376082211
// MI455X (gfx1250) — matrix operand order repaired and checked by reading, not compiled, not run on hardware
//
#include <hip/hip_runtime.h>
#include <hip/hip_bf16.h>

typedef _Float16 v16h __attribute__((ext_vector_type(16)));
typedef _Float16 v8h  __attribute__((ext_vector_type(8)));
typedef float    v8f  __attribute__((ext_vector_type(8)));

#define B_   4096
#define L_   27
#define CIN_ 64
#define H_   256
#define TD_  64
#define LOUT_ 3
#define N3H_ 768   // 3*H

// ws layout (bytes)
#define WVT_OFF 0u            // f16 [768][64]   : WvT, K-contiguous per column
#define WOT_OFF 98304u        // f16 [256][768]  : WoT, K-contiguous per column
#define GH_OFF  491520u       // f16 [4096][768] : gated g, flattened o-major

// ---------------------------------------------------------------------------
// Prep: fp32 weights -> transposed f16 for WMMA B-fragments
// ---------------------------------------------------------------------------
__global__ void prep_weights(const float* __restrict__ Wv,   // [64][768]
                             const float* __restrict__ Wo,   // [768][256]
                             _Float16* __restrict__ wvt,     // [768][64]
                             _Float16* __restrict__ wot) {   // [256][768]
  int idx = blockIdx.x * blockDim.x + threadIdx.x;
  if (idx < CIN_ * N3H_) {
    int k = idx / N3H_, n = idx % N3H_;
    wvt[n * CIN_ + k] = (_Float16)Wv[idx];
  } else {
    int j = idx - CIN_ * N3H_;
    if (j < N3H_ * H_) {
      int k = j / H_, n = j % H_;
      wot[n * N3H_ + k] = (_Float16)Wo[j];
    }
  }
}

// ---------------------------------------------------------------------------
// Main per-batch kernel: LN -> WMMA GEMM (v @ Wv) -> conv kernels -> bmm ->
// target gating -> writes Gh (f16) and out1 partial into d_out
// 256 threads = 8 waves, one block per batch element
// ---------------------------------------------------------------------------
#define VSTRIDE 772  // 27 rows of 768 + pad (bank-conflict dodge)

__global__ void __launch_bounds__(256)
navgraph_main(const float* __restrict__ x,      // [B][27][64]
              const float* __restrict__ target, // [B][64]
              const float* __restrict__ ln_g, const float* __restrict__ ln_b,
              const float* __restrict__ bv,     // [768]
              const float* __restrict__ W0, const float* __restrict__ b0,
              const float* __restrict__ W1, const float* __restrict__ b1,
              const float* __restrict__ W2, const float* __restrict__ b2,
              const float* __restrict__ Wh, const float* __restrict__ bh,
              const float* __restrict__ Wt1, const float* __restrict__ bt1,
              const float* __restrict__ Wt2, const float* __restrict__ bt2,
              const _Float16* __restrict__ wvt, // [768][64]
              _Float16* __restrict__ Gh,        // [B][768]
              float* __restrict__ out)          // [B][256] (out1 partial)
{
  __shared__ _Float16 vnh[32 * 72];        // layernormed x, f16, padded
  __shared__ _Float16 Vh[L_ * VSTRIDE];    // relu(v@Wv+bv), f16
  __shared__ float gls[768];               // g before gating [3][256]
  __shared__ float skl[243];               // sk [3][3][27]
  __shared__ float taS[256];
  __shared__ float h1S[256];
  __shared__ float tgS[64];

  const int b    = blockIdx.x;
  const int tid  = threadIdx.x;
  const int lane = tid & 31;
  const int wid  = tid >> 5;
  const int hi16 = (lane >= 16) ? 1 : 0;
  const int l15  = lane & 15;

  // ---- Stage 1: target to LDS + layernorm rows -> vnh (f16) ----
  if (tid < 64) tgS[tid] = target[b * 64 + tid];
  for (int i = tid; i < 5 * 72; i += 256) vnh[27 * 72 + i] = (_Float16)0.0f;

  for (int row = wid; row < L_; row += 8) {
    const float* xr = x + ((size_t)b * L_ + row) * CIN_;
    float x0 = xr[lane], x1 = xr[lane + 32];
    float s = x0 + x1, sq = x0 * x0 + x1 * x1;
    #pragma unroll
    for (int off = 16; off; off >>= 1) {
      s  += __shfl_xor(s, off);
      sq += __shfl_xor(sq, off);
    }
    float mean = s * (1.0f / 64.0f);
    float var  = sq * (1.0f / 64.0f) - mean * mean;
    float rstd = rsqrtf(var + 1e-5f);
    vnh[row * 72 + lane]      = (_Float16)((x0 - mean) * rstd * ln_g[lane]      + ln_b[lane]);
    vnh[row * 72 + lane + 32] = (_Float16)((x1 - mean) * rstd * ln_g[lane + 32] + ln_b[lane + 32]);
  }
  __syncthreads();

  // ---- Stage 2: WMMA GEMM  vn[32x64] @ Wv[64x768] ----
  // A fragments: [mtile][kchunk], built once per wave, reused over 6 N-tiles
  v16h afr[2][2];
  #pragma unroll
  for (int m = 0; m < 2; ++m) {
    #pragma unroll
    for (int kc = 0; kc < 2; ++kc) {
      const int row = m * 16 + l15;
      const int kadd = kc * 32 + hi16 * 16;
      v16h a;
      #pragma unroll
      for (int e = 0; e < 16; ++e)
        a[e] = vnh[row * 72 + kadd + e];
      afr[m][kc] = a;
    }
  }

  for (int nt = 0; nt < 6; ++nt) {
    const int Ncol = wid * 96 + nt * 16;
    const int N = Ncol + l15;
    const _Float16* bp = wvt + N * CIN_ + hi16 * 16;
    v16h b0f = *(const v16h*)(bp);
    v16h b1f = *(const v16h*)(bp + 32);
    const float bias = bv[N];
    #pragma unroll
    for (int m = 0; m < 2; ++m) {
      v8f acc = {};
      acc = __builtin_amdgcn_wmma_f32_16x16x32_f16(false, afr[m][0], false, b0f,
                                                   (short)0, acc, false, false);
      acc = __builtin_amdgcn_wmma_f32_16x16x32_f16(false, afr[m][1], false, b1f,
                                                   (short)0, acc, false, false);
      #pragma unroll
      for (int r = 0; r < 8; ++r) {
        int M = m * 16 + r + hi16 * 8;
        if (M < L_) {
          float v = acc[r] + bias;
          Vh[M * VSTRIDE + N] = (_Float16)(v > 0.0f ? v : 0.0f);
        }
      }
    }
  }
  __syncthreads();

  // ---- Stage 3: sk[3][3][27] (per-head kernel generators) ----
  for (int task = wid; task < 243; task += 8) {
    const int h = task / 81, rem = task % 81;
    const int o = rem / 27, t = rem % 27;
    float s = 0.0f;
    if (h == 0) {
      for (int d = lane; d < 256; d += 32)
        s += (float)Vh[t * VSTRIDE + d] * W0[d * 3 + o];
    } else {
      const float* Wg = (h == 1) ? W1 : W2;
      const int dil = h;             // 1 or 2
      const int hoff = h * 256;
      for (int i = lane; i < 768; i += 32) {
        int k = i >> 8, c = i & 255;
        int lsrc = (t + dil * k - dil + 54) % 27;
        s += (float)Vh[lsrc * VSTRIDE + hoff + c] * Wg[(o * 256 + c) * 3 + k];
      }
    }
    #pragma unroll
    for (int off = 16; off; off >>= 1) s += __shfl_down(s, off);
    if (lane == 0) {
      float bias = (h == 0) ? b0[o] : ((h == 1) ? b1[o] : b2[o]);
      float v = s + bias;
      skl[task] = v > 0.0f ? v : 0.0f;
    }
  }
  __syncthreads();

  // ---- Stage 4: heads bmm + head_communicate -> gls[3][256] ----
  {
    const float w0h = Wh[0], w1h = Wh[1], w2h = Wh[2], bhh = bh[0];
    #pragma unroll
    for (int i = 0; i < 3; ++i) {
      const int id = tid + i * 256;
      const int o = id >> 8, d = id & 255;
      float g = bhh;
      #pragma unroll
      for (int h = 0; h < 3; ++h) {
        float sh = 0.0f;
        const float wh = (h == 0) ? w0h : ((h == 1) ? w1h : w2h);
        for (int l = 0; l < L_; ++l)
          sh += skl[h * 81 + o * 27 + l] * (float)Vh[l * VSTRIDE + h * 256 + d];
        g += sh * wh;
      }
      gls[id] = g;
    }
  }
  __syncthreads();

  // ---- Stage 5: target gating MLP ----
  {
    float s = bt1[tid];
    for (int k = 0; k < 64; ++k) s += tgS[k] * Wt1[k * 256 + tid];
    h1S[tid] = s > 0.0f ? s : 0.0f;
  }
  __syncthreads();
  {
    float s = bt2[tid];
    for (int j = 0; j < 256; ++j) s += h1S[j] * Wt2[j * 256 + tid];
    taS[tid] = s > 0.0f ? s : 0.0f;
  }
  __syncthreads();

  // ---- Stage 6: gate, avg-pool partial to out, f16 flat g to Gh ----
  {
    const int d = tid;
    const float tav = taS[d];
    float g0 = gls[d]       * tav;
    float g1 = gls[256 + d] * tav;
    float g2 = gls[512 + d] * tav;
    _Float16* gp = Gh + (size_t)b * 768;
    gp[d]       = (_Float16)g0;
    gp[256 + d] = (_Float16)g1;
    gp[512 + d] = (_Float16)g2;
    out[(size_t)b * 256 + d] = (g0 + g1 + g2) * (1.0f / 3.0f);
  }
}

// ---------------------------------------------------------------------------
// Tail GEMM: out += relu(Gh[4096,768] @ Wo[768,256] + bo)
// 8 waves / block, one 16x16 tile per wave, K loop = 24 wmma
// ---------------------------------------------------------------------------
__global__ void __launch_bounds__(256)
tail_gemm(const _Float16* __restrict__ Gh,   // [4096][768]
          const _Float16* __restrict__ wot,  // [256][768]
          const float* __restrict__ bo,      // [256]
          float* __restrict__ out)           // [4096][256]
{
  const int tid  = threadIdx.x;
  const int lane = tid & 31;
  const int wid  = tid >> 5;
  const int hi16 = (lane >= 16) ? 1 : 0;
  const int l15  = lane & 15;

  const int tile = blockIdx.x * 8 + wid;     // 0..4095
  const int tm = tile >> 4;                  // 0..255 (M tiles of 16)
  const int tn = tile & 15;                  // 0..15  (N tiles of 16)

  const int row  = tm * 16 + l15;
  const int nIdx = tn * 16 + l15;
  const _Float16* arow = Gh  + (size_t)row * 768 + hi16 * 16;
  const _Float16* bcol = wot + (size_t)nIdx * 768 + hi16 * 16;

  v8f acc = {};
  #pragma unroll 4
  for (int kc = 0; kc < 24; ++kc) {
    v8h lo = *(const v8h*)(arow + kc * 32);
    v8h hi = *(const v8h*)(arow + kc * 32 + 8);
    v16h a;
    #pragma unroll
    for (int e = 0; e < 8; ++e) { a[e] = lo[e]; a[e + 8] = hi[e]; }
    v16h bb = *(const v16h*)(bcol + kc * 32);
    acc = __builtin_amdgcn_wmma_f32_16x16x32_f16(false, a, false, bb,
                                                 (short)0, acc, false, false);
  }

  const int N = tn * 16 + l15;
  const float bias = bo[N];
  #pragma unroll
  for (int r = 0; r < 8; ++r) {
    const int M = tm * 16 + r + hi16 * 8;
    float v = acc[r] + bias;
    v = v > 0.0f ? v : 0.0f;
    float* op = out + (size_t)M * 256 + N;
    *op = v + *op;   // add avg-pool residual written by navgraph_main
  }
}

// ---------------------------------------------------------------------------
extern "C" void kernel_launch(void* const* d_in, const int* in_sizes, int n_in,
                              void* d_out, int out_size, void* d_ws, size_t ws_size,
                              hipStream_t stream) {
  const float* x      = (const float*)d_in[0];
  const float* target = (const float*)d_in[1];
  const float* ln_g   = (const float*)d_in[2];
  const float* ln_b   = (const float*)d_in[3];
  const float* Wv     = (const float*)d_in[4];
  const float* bv     = (const float*)d_in[5];
  const float* W0     = (const float*)d_in[6];
  const float* b0     = (const float*)d_in[7];
  const float* W1     = (const float*)d_in[8];
  const float* b1     = (const float*)d_in[9];
  const float* W2     = (const float*)d_in[10];
  const float* b2     = (const float*)d_in[11];
  const float* Wh     = (const float*)d_in[12];
  const float* bh     = (const float*)d_in[13];
  const float* Wt1    = (const float*)d_in[14];
  const float* bt1    = (const float*)d_in[15];
  const float* Wt2    = (const float*)d_in[16];
  const float* bt2    = (const float*)d_in[17];
  const float* Wo     = (const float*)d_in[18];
  const float* bo     = (const float*)d_in[19];

  float* out = (float*)d_out;
  _Float16* wvt = (_Float16*)((char*)d_ws + WVT_OFF);
  _Float16* wot = (_Float16*)((char*)d_ws + WOT_OFF);
  _Float16* Gh  = (_Float16*)((char*)d_ws + GH_OFF);

  // 1) convert weights to transposed f16 (49152 + 196608 elements)
  {
    int total = CIN_ * N3H_ + N3H_ * H_;
    int blocks = (total + 255) / 256;
    prep_weights<<<blocks, 256, 0, stream>>>(Wv, Wo, wvt, wot);
  }

  // 2) main per-batch pipeline (WMMA GEMM + small ops)
  navgraph_main<<<B_, 256, 0, stream>>>(x, target, ln_g, ln_b, bv,
                                        W0, b0, W1, b1, W2, b2, Wh, bh,
                                        Wt1, bt1, Wt2, bt2,
                                        wvt, Gh, out);

  // 3) tail batched GEMM with residual add (4096 tiles / 8 waves per block)
  tail_gemm<<<512, 256, 0, stream>>>(Gh, wot, bo, out);
}
